// RelativeSAMAttention_40973988004737
// MI455X (gfx1250) — matrix operand order repaired and checked by reading, not compiled, not run on hardware
//
#include <hip/hip_runtime.h>
#include <hip/hip_bf16.h>
#include <math.h>

typedef __attribute__((ext_vector_type(16))) _Float16 v16h;
typedef __attribute__((ext_vector_type(8)))  _Float16 v8h;
typedef __attribute__((ext_vector_type(8)))  float    v8f;

#define BB 2
#define HH 8
#define NN 1024
#define MM 1024
#define DM 8
#define LC 16
#define DIN 18

// workspace layout (float offsets)
#define WS_MVAL 0
#define WS_SP   8                  // 16 floats: softplus sums per (b,h)
#define WS_DETR 32                 // B*H*N floats: row sums of relu scores
#define WS_PART (32 + BB*HH*NN)    // 4096 floats: block partials for m-reduction

// Branchless exact-GELU: erf via Abramowitz-Stegun 7.1.26 (|err| < 1.5e-7).
// Uses native v_rcp_f32 / v_exp_f32 so no libm branches / EXEC juggling.
__device__ __forceinline__ float gelu_exact(float x) {
  float z  = x * 0.70710678118654752f;
  float az = __builtin_fabsf(z);
  float t  = __builtin_amdgcn_rcpf(__builtin_fmaf(0.3275911f, az, 1.f));
  float p  = __builtin_fmaf(t, 1.061405429f, -1.453152027f);
  p = __builtin_fmaf(t, p, 1.421413741f);
  p = __builtin_fmaf(t, p, -0.284496736f);
  p = __builtin_fmaf(t, p, 0.254829592f);
  p = p * t;
  float e  = __builtin_amdgcn_exp2f(-az * az * 1.4426950408889634f);
  float er = __builtin_fmaf(-p, e, 1.f);           // erf(|z|)
  er = __builtin_copysignf(er, z);
  return 0.5f * x * (1.f + er);
}

// ---------------------------------------------------------------------------
// Kernel 1: per-block partial sums of min_b(pairwise sq dist) over (n,m)
// ---------------------------------------------------------------------------
__global__ __launch_bounds__(256) void sam_dmin(
    const float* __restrict__ dq, const float* __restrict__ dk,
    float* __restrict__ part)
{
  const int t = blockIdx.x * 256 + threadIdx.x;  // 0 .. N*M-1
  const int n = t >> 10;
  const int m = t & (MM - 1);
  float best = 3.4e38f;
  #pragma unroll
  for (int b = 0; b < BB; ++b) {
    const float* a = dq + (size_t)(b * NN + n) * 6;
    const float* c = dk + (size_t)(b * MM + m) * 6;
    float xn = a[0]*a[0] + a[1]*a[1] + a[2]*a[2];
    float yn = c[0]*c[0] + c[1]*c[1] + c[2]*c[2];
    float dt = a[0]*c[0] + a[1]*c[1] + a[2]*c[2];
    best = fminf(best, xn + yn - 2.f * dt);   // matches reference formula
  }
  __shared__ float red[256];
  red[threadIdx.x] = best;
  __syncthreads();
  for (int s = 128; s > 0; s >>= 1) {
    if (threadIdx.x < s) red[threadIdx.x] += red[threadIdx.x + s];
    __syncthreads();
  }
  if (threadIdx.x == 0) part[blockIdx.x] = red[0];
}

// ---------------------------------------------------------------------------
// Kernel 2: finalize m scalar + softplus sums (single block, deterministic)
// ---------------------------------------------------------------------------
__global__ __launch_bounds__(256) void sam_finalize(
    const float* __restrict__ part, const float* __restrict__ c,
    float* __restrict__ wsf)
{
  if (threadIdx.x < BB * HH) {
    const float* cp = c + threadIdx.x * LC;
    float sp = 0.f;
    #pragma unroll
    for (int l = 0; l < LC; ++l) {
      float x = cp[l];
      sp += (x > 20.f) ? x : log1pf(expf(x));   // stable softplus
    }
    wsf[WS_SP + threadIdx.x] = sp;
  }
  float s = 0.f;
  for (int i = threadIdx.x; i < 4096; i += 256) s += part[i];
  __shared__ float red[256];
  red[threadIdx.x] = s;
  __syncthreads();
  for (int st = 128; st > 0; st >>= 1) {
    if (threadIdx.x < st) red[threadIdx.x] += red[threadIdx.x + st];
    __syncthreads();
  }
  if (threadIdx.x == 0) wsf[WS_MVAL] = red[0] / (float)(NN * MM);
}

// ---------------------------------------------------------------------------
// Kernel 3: WMMA MLP + LayerNorm + ReLU. One wave per (b,n); 64 m-tiles of 16
// pixels. Writes UNNORMALIZED scores and per-(b,h,n) row sums.
//
// v_wmma_f32_16x16x32_f16 operand layouts used (CDNA5 ISA):
//   A (16xK=32 f16): lanes 0-15 = rows, elems 0..7 -> K=8*half+j,
//                    elems 8..15 -> K=16+8*half+(j-8)
//   B (K=32 x 16 f16): lane%16 = N col, elem j -> same K as A (8*half+j / 16+8*half+(j-8))
//   C/D (16x16 f32): lane%16 = N col, reg r -> row r + 8*half
// ---------------------------------------------------------------------------
__global__ __launch_bounds__(256) void sam_pass1(
    const float* __restrict__ q,  const float* __restrict__ k,
    const float* __restrict__ dq, const float* __restrict__ dk,
    const float* __restrict__ W1, const float* __restrict__ b1,
    const float* __restrict__ W2, const float* __restrict__ b2,
    const float* __restrict__ Wl, const float* __restrict__ bl,
    const float* __restrict__ gamma, const float* __restrict__ beta,
    const float* __restrict__ wsf, float* __restrict__ detr,
    float* __restrict__ scores)
{
  // per-wave 16x32 f16 staging tile; row stride 40 halves (80B, keeps the
  // 8-half (16B) ds_load_b128 reads aligned)
  __shared__ __align__(16) _Float16 h1buf[8][16 * 40];

  const int lane = threadIdx.x & 31;
  const int wid  = threadIdx.x >> 5;
  const int gw   = blockIdx.x * 8 + wid;   // global wave = (b, n)
  const int b    = gw >> 10;
  const int n    = gw & (NN - 1);
  const int half = lane >> 4;
  const int sub  = lane & 15;
  _Float16* h1 = h1buf[wid];

  // ---- constant B fragments -------------------------------------------------
  v16h fW1a, fW1b, fW2, fWl;
  #pragma unroll
  for (int j = 0; j < 16; ++j) {
    int K = (j < 8) ? (8 * half + j) : (16 + 8 * half + (j - 8));
    fW1a[j] = (K < DIN) ? (_Float16)W1[K * 32 + sub]       : (_Float16)0.f;
    fW1b[j] = (K < DIN) ? (_Float16)W1[K * 32 + 16 + sub]  : (_Float16)0.f;
    fW2[j]  = (sub < HH) ? (_Float16)W2[K * 8 + sub]       : (_Float16)0.f;
    fWl[j]  = (K < DIN && sub < HH) ? (_Float16)Wl[K * 8 + sub] : (_Float16)0.f;
  }
  const float b1lo = b1[sub];
  const float b1hi = b1[16 + sub];
  const float cb   = (sub < HH) ? (b2[sub] + bl[sub]) : 0.f;
  const float gmm  = (sub < HH) ? gamma[sub] : 0.f;
  const float bet  = (sub < HH) ? beta[sub]  : 0.f;

  // loop-invariant C fragments (biases)
  v8f c0, c1, cl;
  #pragma unroll
  for (int r = 0; r < 8; ++r) { c0[r] = b1lo; c1[r] = b1hi; cl[r] = cb; }

  const float mval = wsf[WS_MVAL];
  const float invm = 1.f / mval;

  const float* dqp = dq + (size_t)(b * NN + n) * 6;
  const float qx = dqp[0], qy = dqp[1], qz = dqp[2];
  const float qn3 = qx * qx + qy * qy + qz * qz;

  // ---- base A fragment: K=0..7 are the 8 q-features, constant over m -------
  v16h aF;
  #pragma unroll
  for (int j = 0; j < 16; ++j) aF[j] = (_Float16)0.f;
  if (half == 0) {
    #pragma unroll
    for (int h = 0; h < HH; ++h)
      aF[h] = (_Float16)q[(size_t)(b * HH + h) * NN + n];
  }

  const float inv_sqrt3 = 0.5773502691896258f;
  float dacc = 0.f;   // running sum of relu scores for this lane's h column

  // ---- software pipeline: prefetch tile-0 operand data ----------------------
  float pkx = 0.f, pky = 0.f, pkz = 0.f;
  float pk[HH];
  if (half == 0) {
    const float* dkp = dk + (size_t)(b * MM + sub) * 6;
    pkx = dkp[0]; pky = dkp[1]; pkz = dkp[2];
  } else {
    #pragma unroll
    for (int h = 0; h < HH; ++h) pk[h] = k[(size_t)(b * HH + h) * MM + sub];
  }

  for (int mt = 0; mt < MM / 16; ++mt) {
    const int m0 = mt * 16;

    // ---- build A fragment from prefetched data -----------------------------
    if (half == 0) {   // lanes 0-15: geometry features for pixel m0+sub
      float dot  = qx * pkx + qy * pky + qz * pkz;
      float kn3  = pkx * pkx + pky * pky + pkz * pkz;
      float dist = qn3 + kn3 - 2.f * dot;
      aF[8] = (_Float16)((dist - mval) * invm);  // K=16: dmap_n
      aF[9] = (_Float16)(dot * inv_sqrt3);       // K=17: nmap
    } else {           // lanes 16-31: K=8..15 are the 8 k-features
      #pragma unroll
      for (int h = 0; h < HH; ++h) aF[h] = (_Float16)pk[h];
    }

    // ---- issue next tile's loads now (clamped); consumed next iteration ----
    {
      int mr2 = m0 + 16 + sub;
      if (mr2 > MM - 1) mr2 = MM - 1;
      if (half == 0) {
        const float* dkp = dk + (size_t)(b * MM + mr2) * 6;
        pkx = dkp[0]; pky = dkp[1]; pkz = dkp[2];
      } else {
        #pragma unroll
        for (int h = 0; h < HH; ++h) pk[h] = k[(size_t)(b * HH + h) * MM + mr2];
      }
    }

    // ---- GEMM-1: hidden = a @ W1 + b1 (32 cols = 2 WMMAs) ------------------
    v8f d0 = __builtin_amdgcn_wmma_f32_16x16x32_f16(false, aF, false, fW1a, (short)0, c0, false, false);
    v8f d1 = __builtin_amdgcn_wmma_f32_16x16x32_f16(false, aF, false, fW1b, (short)0, c1, false, false);

    // ---- branchless exact GELU, stage to LDS in A-matrix-readable form -----
    #pragma unroll
    for (int r = 0; r < 8; ++r) {
      int row = r + 8 * half;
      h1[row * 40 + sub]      = (_Float16)gelu_exact(d0[r]);
      h1[row * 40 + 16 + sub] = (_Float16)gelu_exact(d1[r]);
    }
    __builtin_amdgcn_wave_barrier();  // same-wave LDS ops are in-order; keep issue order
    const v8h lo = *(const v8h*)&h1[sub * 40 + 8 * half];
    const v8h hi = *(const v8h*)&h1[sub * 40 + 16 + 8 * half];
    v16h aH = __builtin_shufflevector(lo, hi, 0,1,2,3,4,5,6,7,8,9,10,11,12,13,14,15);
    __builtin_amdgcn_wave_barrier();

    // ---- GEMM-2 chained with the linear path -------------------------------
    v8f dl = __builtin_amdgcn_wmma_f32_16x16x32_f16(false, aF, false, fWl, (short)0, cl, false, false);
    v8f d2 = __builtin_amdgcn_wmma_f32_16x16x32_f16(false, aH, false, fW2, (short)0, dl, false, false);

    // ---- LayerNorm over H=8 (xor 1/2/4 stays inside each 16-lane half) -----
    float sv[8];
    #pragma unroll
    for (int r = 0; r < 8; ++r) {
      float x  = d2[r];
      float s1 = x, s2 = x * x;
      #pragma unroll
      for (int msk = 1; msk <= 4; msk <<= 1) {
        s1 += __shfl_xor(s1, msk, 32);
        s2 += __shfl_xor(s2, msk, 32);
      }
      float mu  = s1 * 0.125f;
      float var = s2 * 0.125f - mu * mu;
      float hn  = (x - mu) * __builtin_amdgcn_rsqf(var + 1e-5f);
      float s   = fmaxf(hn * gmm + bet, 0.f);
      sv[r] = s;
      dacc += s;
    }

    // ---- write unnormalized scores: lane col = h, rows = 8 consecutive m ---
    if (sub < HH) {
      float4 lo4 = make_float4(sv[0], sv[1], sv[2], sv[3]);
      float4 hi4 = make_float4(sv[4], sv[5], sv[6], sv[7]);
      float* dst = scores + ((size_t)(b * HH + sub) * NN + n) * MM + m0 + 8 * half;
      *(float4*)(dst)     = lo4;
      *(float4*)(dst + 4) = hi4;
    }
  }

  dacc += __shfl_xor(dacc, 16, 32);   // combine the two half-rows of each h
  if (half == 0 && sub < HH)
    detr[(size_t)(b * HH + sub) * NN + n] = dacc;
}

// ---------------------------------------------------------------------------
// Kernel 4: normalize scores in place and out = scores @ v (v cached in LDS)
// ---------------------------------------------------------------------------
__global__ __launch_bounds__(256) void sam_pass2(
    const float* __restrict__ v, const float* __restrict__ wsf,
    float* __restrict__ scores, float* __restrict__ out)
{
  __shared__ float vt[MM * DM];           // 32 KB: v[b,h,:,:]
  const int bh  = blockIdx.x >> 5;
  const int seg = blockIdx.x & 31;
  const float* vp = v + (size_t)bh * MM * DM;
  for (int i = threadIdx.x; i < MM * DM; i += 256) vt[i] = vp[i];
  __syncthreads();

  const int lane = threadIdx.x & 31;
  const int wid  = threadIdx.x >> 5;
  const float sp = wsf[WS_SP + bh];

  for (int rr = 0; rr < 4; ++rr) {
    const int n = seg * 32 + wid * 4 + rr;
    const float inv = 1.f / (wsf[WS_DETR + bh * NN + n] + sp + 1e-9f);
    float acc[DM];
    #pragma unroll
    for (int d = 0; d < DM; ++d) acc[d] = 0.f;
    float* srow = scores + ((size_t)bh * NN + n) * MM;
    for (int m = lane; m < MM; m += 32) {
      float s = srow[m] * inv;
      srow[m] = s;                        // normalized scores output
      #pragma unroll
      for (int d = 0; d < DM; ++d) acc[d] += s * vt[m * DM + d];
    }
    #pragma unroll
    for (int d = 0; d < DM; ++d) {
      #pragma unroll
      for (int msk = 16; msk >= 1; msk >>= 1)
        acc[d] += __shfl_xor(acc[d], msk, 32);
    }
    if (lane == 0) {
      #pragma unroll
      for (int d = 0; d < DM; ++d)
        out[((size_t)bh * NN + n) * DM + d] = acc[d];
    }
  }
}

// ---------------------------------------------------------------------------
extern "C" void kernel_launch(void* const* d_in, const int* in_sizes, int n_in,
                              void* d_out, int out_size, void* d_ws, size_t ws_size,
                              hipStream_t stream) {
  const float* q  = (const float*)d_in[0];
  const float* k  = (const float*)d_in[1];
  const float* v  = (const float*)d_in[2];
  const float* c  = (const float*)d_in[3];
  const float* dq = (const float*)d_in[4];
  const float* dk = (const float*)d_in[5];
  const float* W1 = (const float*)d_in[6];
  const float* b1 = (const float*)d_in[7];
  const float* W2 = (const float*)d_in[8];
  const float* b2 = (const float*)d_in[9];
  const float* Wl = (const float*)d_in[10];
  const float* bl = (const float*)d_in[11];
  const float* gm = (const float*)d_in[12];
  const float* bt = (const float*)d_in[13];

  float* out    = (float*)d_out;                        // (B,H,N,Dm)
  float* scores = out + (size_t)BB * HH * NN * DM;      // (B,H,N,M)
  float* wsf    = (float*)d_ws;                         // ~82 KB used

  sam_dmin    <<<dim3((NN * MM) / 256), dim3(256), 0, stream>>>(dq, dk, wsf + WS_PART);
  sam_finalize<<<dim3(1),               dim3(256), 0, stream>>>(wsf + WS_PART, c, wsf);
  sam_pass1   <<<dim3((BB * NN) / 8),   dim3(256), 0, stream>>>(q, k, dq, dk, W1, b1, W2, b2,
                                                                Wl, bl, gm, bt, wsf,
                                                                wsf + WS_DETR, scores);
  sam_pass2   <<<dim3(BB * HH * 32),    dim3(256), 0, stream>>>(v, wsf, scores, out);
}
